// Seq2Seq_37993280700829
// MI455X (gfx1250) — matrix operand order repaired and checked by reading, not compiled, not run on hardware
//
#include <hip/hip_runtime.h>
#include <hip/hip_bf16.h>

typedef __attribute__((ext_vector_type(16))) __bf16 v16bf;
typedef __attribute__((ext_vector_type(8)))  __bf16 v8bf;
typedef __attribute__((ext_vector_type(8)))  float  v8f;

#define HID      1024
#define BATCH    32
#define SRCLEN   64
#define TSTEPS   15
#define VOCAB    58334
#define VOCABP   58336      // padded to multiple of 16
#define EK       256        // encoder input K (250 padded)
#define DK       1280       // decoder input K (1274 padded)
#define GATES4H  4096
#define RBLOCKS  8          // persistent recurrence grid
#define NBARS    (SRCLEN + 3 * TSTEPS)   // 109 grid barriers

// ---------------- WMMA fragment helpers (bf16 16x16x32, per ISA layout tables) ------------

// A-matrix 16x32 bf16: lanes 0-15 -> M=0..15, K 0..7 (v0..3) & 16..23 (v4..7);
// lanes 16-31 -> same M, K 8..15 & 24..31.
__device__ __forceinline__ v16bf load_a_frag(const __bf16* __restrict__ A, int lda,
                                             int m0, int k0, int lane) {
  int m  = m0 + (lane & 15);
  int kh = (lane >> 4) * 8;
  const __bf16* p = A + (size_t)m * lda + k0 + kh;
  v8bf lo = *(const v8bf*)(p);
  v8bf hi = *(const v8bf*)(p + 16);
  v16bf r;
#pragma unroll
  for (int i = 0; i < 8; ++i) { r[i] = lo[i]; r[8 + i] = hi[i]; }
  return r;
}

// B-matrix 32x16 bf16 with B[k][n] = W[n][k] (W row-major [N,K]):
// lanes 0-15 -> col n, K 0..7 & 16..23; lanes 16-31 -> col n, K 8..15 & 24..31.
__device__ __forceinline__ v16bf load_b_frag(const __bf16* __restrict__ W, int ldw,
                                             int n0, int k0, int lane) {
  int n  = n0 + (lane & 15);
  int kh = (lane >> 4) * 8;
  const __bf16* p = W + (size_t)n * ldw + k0 + kh;
  v8bf lo = *(const v8bf*)(p);
  v8bf hi = *(const v8bf*)(p + 16);
  v16bf r;
#pragma unroll
  for (int i = 0; i < 8; ++i) { r[i] = lo[i]; r[8 + i] = hi[i]; }
  return r;
}

__device__ __forceinline__ v8f wmma_bf16(v16bf a, v16bf b, v8f c) {
  return __builtin_amdgcn_wmma_f32_16x16x32_bf16(false, a, false, b, (short)0, c, false, false);
}

__device__ __forceinline__ float sigmoidf_(float x) { return 1.f / (1.f + __expf(-x)); }

// ---------------- software grid barrier (8 co-resident blocks, fresh counter per slot) ----

__device__ __forceinline__ void grid_barrier(unsigned int* cnt) {
  __syncthreads();
  if (threadIdx.x == 0) {
    __threadfence();
    atomicAdd(cnt, 1u);
    while (atomicAdd(cnt, 0u) < (unsigned)RBLOCKS) { __builtin_amdgcn_s_sleep(1); }
  }
  __syncthreads();
  __threadfence();
}

// ---------------- utility kernels ----------------

__global__ void zero_u32_kernel(unsigned int* __restrict__ p, int n) {
  int i = blockIdx.x * blockDim.x + threadIdx.x;
  if (i < n) p[i] = 0u;
}

// f32 [srows,scols] -> bf16 [drows,dcols], zero padded
__global__ void cvt_pad_kernel(const float* __restrict__ src, __bf16* __restrict__ dst,
                               int srows, int scols, int drows, int dcols) {
  long long i = (long long)blockIdx.x * blockDim.x + threadIdx.x;
  long long n = (long long)drows * dcols;
  if (i >= n) return;
  int r = (int)(i / dcols), c = (int)(i % dcols);
  float v = (r < srows && c < scols) ? src[(size_t)r * scols + c] : 0.f;
  dst[i] = (__bf16)v;
}

// src token embeddings -> bf16 [SRCLEN][BATCH][EK]
__global__ void src_embed_kernel(const int* __restrict__ src, const float* __restrict__ emb,
                                 __bf16* __restrict__ x) {
  int i = blockIdx.x * blockDim.x + threadIdx.x;
  if (i >= SRCLEN * BATCH * EK) return;
  int t = i / (BATCH * EK);
  int r = i % (BATCH * EK);
  int b = r / EK, k = r % EK;
  int tok = src[b * SRCLEN + t];
  x[i] = (k < 250) ? (__bf16)emb[(size_t)tok * 250 + k] : (__bf16)0.f;
}

// ---------------- LSTM tile: gates GEMM (WMMA) + in-register c/h update ------------------
// One wave owns a 16-wide j-tile: all four gates, M=32 rows, 8 accumulators.
// k-loops at unroll 2: iteration i+1's 12 b128 loads overlap iteration i's 8 WMMAs
// (unroll 1 drained loadcnt to 0 before every WMMA group; full unroll spilled).
// Live regs: 2x6 fragments (96) + 64 accums + addressing ~ 200 VGPRs, no spill.
template <int KX>
__device__ __forceinline__ void lstm_tile(
    const __bf16* __restrict__ x,
    const __bf16* __restrict__ Wih,          // [4096, KX] bf16
    const __bf16* __restrict__ Whh,          // [4096, 1024] bf16
    const float*  __restrict__ bias,         // [4096] f32
    const __bf16* __restrict__ h_in,         // [32,1024] bf16
    float* __restrict__ c,                   // [32,1024] f32 in/out (tile-private)
    float* __restrict__ h_out_f,             // [32,1024] f32
    __bf16* __restrict__ h_out_bf,           // [32,1024] bf16
    float* __restrict__ enc_out_t,           // may be null; index b*SRCLEN*HID + j
    __bf16* __restrict__ hist,               // may be null; [32,1024] bf16 history slot
    int jt, int lane)
{
  int j0 = jt * 16;
  v8f acc[4][2] = {};                        // [gate][m-tile]

#pragma unroll 2
  for (int k0 = 0; k0 < KX; k0 += 32) {
    v16bf a0 = load_a_frag(x, KX, 0, k0, lane);
    v16bf a1 = load_a_frag(x, KX, 16, k0, lane);
#pragma unroll
    for (int g = 0; g < 4; ++g) {
      v16bf bf = load_b_frag(Wih, KX, g * HID + j0, k0, lane);
      acc[g][0] = wmma_bf16(a0, bf, acc[g][0]);
      acc[g][1] = wmma_bf16(a1, bf, acc[g][1]);
    }
  }
#pragma unroll 2
  for (int k0 = 0; k0 < HID; k0 += 32) {
    v16bf a0 = load_a_frag(h_in, HID, 0, k0, lane);
    v16bf a1 = load_a_frag(h_in, HID, 16, k0, lane);
#pragma unroll
    for (int g = 0; g < 4; ++g) {
      v16bf bf = load_b_frag(Whh, HID, g * HID + j0, k0, lane);
      acc[g][0] = wmma_bf16(a0, bf, acc[g][0]);
      acc[g][1] = wmma_bf16(a1, bf, acc[g][1]);
    }
  }

  int j = j0 + (lane & 15);
  int rbase = (lane >> 4) * 8;
  float bi = bias[j], bff = bias[HID + j], bg = bias[2 * HID + j], bo = bias[3 * HID + j];
#pragma unroll
  for (int mt = 0; mt < 2; ++mt) {
#pragma unroll
    for (int r = 0; r < 8; ++r) {
      int b = mt * 16 + rbase + r;
      float iv = sigmoidf_(acc[0][mt][r] + bi);
      float fv = sigmoidf_(acc[1][mt][r] + bff);
      float gv = tanhf(acc[2][mt][r] + bg);
      float ov = sigmoidf_(acc[3][mt][r] + bo);
      size_t idx = (size_t)b * HID + j;
      float cv = fv * c[idx] + iv * gv;
      c[idx] = cv;
      float hv = ov * tanhf(cv);
      h_out_f[idx]  = hv;
      h_out_bf[idx] = (__bf16)hv;
      if (enc_out_t) enc_out_t[(size_t)b * SRCLEN * HID + j] = hv;
      if (hist)      hist[idx] = (__bf16)hv;
    }
  }
}

// ---------------- persistent recurrence kernel: 64 enc steps + 15 dec steps --------------
__global__ void __launch_bounds__(256) recurrence_kernel(
    const __bf16* __restrict__ srcEmb,       // [SRCLEN][BATCH][EK]
    const __bf16* __restrict__ encWih, const __bf16* __restrict__ encWhh,
    const float*  __restrict__ enc_b,
    const __bf16* __restrict__ decWih, const __bf16* __restrict__ decWhh,
    const float*  __restrict__ dec_b,
    const int*    __restrict__ target, const float* __restrict__ emb,
    __bf16* __restrict__ hbf0, __bf16* __restrict__ hbf1,
    float*  __restrict__ hF,   float* __restrict__ cF, float* __restrict__ ctx,
    __bf16* __restrict__ xdec, __bf16* __restrict__ hist,
    float*  __restrict__ encOut, unsigned int* __restrict__ barcnt)
{
  __shared__ float part[256];
  __shared__ float sc[SRCLEN];
  __shared__ float invz;

  int wave = threadIdx.x >> 5;
  int lane = threadIdx.x & 31;
  int jt   = blockIdx.x * 8 + wave;          // 0..63, exact
  int bar  = 0;

  // ---- encoder ----
#pragma unroll 1
  for (int t = 0; t < SRCLEN; ++t) {
    const __bf16* hin = (t & 1) ? hbf1 : hbf0;
    __bf16* hout      = (t & 1) ? hbf0 : hbf1;
    lstm_tile<EK>(srcEmb + (size_t)t * BATCH * EK, encWih, encWhh, enc_b,
                  hin, cF, hF, hout, encOut + (size_t)t * HID, (__bf16*)nullptr, jt, lane);
    grid_barrier(barcnt + (bar++));
  }

  // ---- decoder ----
#pragma unroll 1
  for (int d = 0; d < TSTEPS; ++d) {
    int t = SRCLEN + d;
    // phase A: build x = [emb(tok) | ctx] bf16
    for (int i = blockIdx.x * 256 + threadIdx.x; i < BATCH * DK; i += RBLOCKS * 256) {
      int b = i / DK, k = i % DK;
      int tok = (d == 0) ? 1 : target[b * TSTEPS + (d - 1)];
      float v;
      if (k < 250)       v = emb[(size_t)tok * 250 + k];
      else if (k < 1274) v = ctx[b * HID + (k - 250)];
      else               v = 0.f;
      xdec[i] = (__bf16)v;
    }
    grid_barrier(barcnt + (bar++));

    // phase B: LSTM step
    const __bf16* hin = (t & 1) ? hbf1 : hbf0;
    __bf16* hout      = (t & 1) ? hbf0 : hbf1;
    lstm_tile<DK>(xdec, decWih, decWhh, dec_b, hin, cF, hF, hout,
                  (float*)nullptr, hist + (size_t)d * BATCH * HID, jt, lane);
    grid_barrier(barcnt + (bar++));

    // phase C: attention for 4 batch rows per block
#pragma unroll 1
    for (int bb = 0; bb < 4; ++bb) {
      int b = blockIdx.x * 4 + bb;
      const float* eb = encOut + (size_t)b * SRCLEN * HID;
      const float* hb = hF + (size_t)b * HID;
      int tid = threadIdx.x;
      int s = tid >> 2, q = tid & 3;
      float sum = 0.f;
      for (int j = q * 256; j < q * 256 + 256; ++j) sum += hb[j] * eb[(size_t)s * HID + j];
      part[tid] = sum;
      __syncthreads();
      if (q == 0) sc[s] = part[tid] + part[tid + 1] + part[tid + 2] + part[tid + 3];
      __syncthreads();
      if (tid == 0) {
        float m = sc[0];
        for (int i = 1; i < SRCLEN; ++i) m = fmaxf(m, sc[i]);
        float z = 0.f;
        for (int i = 0; i < SRCLEN; ++i) { sc[i] = __expf(sc[i] - m); z += sc[i]; }
        invz = 1.f / z;
      }
      __syncthreads();
      float inv = invz;
      for (int j = tid; j < HID; j += 256) {
        float a = 0.f;
        for (int s2 = 0; s2 < SRCLEN; ++s2) a += sc[s2] * eb[(size_t)s2 * HID + j];
        ctx[(size_t)b * HID + j] = a * inv;
      }
      __syncthreads();
    }
    grid_barrier(barcnt + (bar++));
  }
}

// ---------------- batched output projection: [480,1024] x [1024,58336] + bias ------------
// Each wave: 1 N-tile, 10 M-tiles (160 rows) -> out_W fragments amortized 10x.
// unroll 1: 80 accum VGPRs leave no headroom for a second iteration's fragments;
// latency hidden by occupancy (10938 waves) instead.
__global__ void __launch_bounds__(256) logits_batched_kernel(
    const __bf16* __restrict__ hist,   // [TSTEPS*BATCH, 1024] bf16, row = d*32+b
    const __bf16* __restrict__ Wo,     // [VOCABP, 1024] bf16
    const float*  __restrict__ ob,
    float* __restrict__ out)           // [BATCH][TSTEPS][VOCAB]
{
  int wave = threadIdx.x >> 5;
  int lane = threadIdx.x & 31;
  int w = blockIdx.x * 8 + wave;
  if (w >= (VOCABP / 16) * 3) return;        // wave-uniform
  int g  = w % 3;                            // row group: rows g*160 .. +159
  int nt = w / 3;
  int n0 = nt * 16;
  int m0 = g * 160;

  v8f acc[10] = {};
#pragma unroll 1
  for (int k0 = 0; k0 < HID; k0 += 32) {
    v16bf bf = load_b_frag(Wo, HID, n0, k0, lane);
#pragma unroll
    for (int mt = 0; mt < 10; ++mt) {
      v16bf a = load_a_frag(hist, HID, m0 + mt * 16, k0, lane);
      acc[mt] = wmma_bf16(a, bf, acc[mt]);
    }
  }

  int v = n0 + (lane & 15);
  if (v < VOCAB) {
    float bb = ob[v];
    int rbase = (lane >> 4) * 8;
#pragma unroll
    for (int mt = 0; mt < 10; ++mt) {
#pragma unroll
      for (int r = 0; r < 8; ++r) {
        int row = m0 + mt * 16 + rbase + r;  // 0..479
        int d = row >> 5, b = row & 31;
        out[((size_t)b * TSTEPS + d) * VOCAB + v] = acc[mt][r] + bb;
      }
    }
  }
}

// ---------------- host-side orchestration ----------------

extern "C" void kernel_launch(void* const* d_in, const int* in_sizes, int n_in,
                              void* d_out, int out_size, void* d_ws, size_t ws_size,
                              hipStream_t stream) {
  (void)in_sizes; (void)n_in; (void)out_size; (void)ws_size;
  const int*   src     = (const int*)  d_in[0];
  const int*   target  = (const int*)  d_in[1];
  const float* emb     = (const float*)d_in[4];
  const float* encWih  = (const float*)d_in[5];
  const float* encWhh  = (const float*)d_in[6];
  const float* enc_b   = (const float*)d_in[7];
  const float* decWih  = (const float*)d_in[8];
  const float* decWhh  = (const float*)d_in[9];
  const float* dec_b   = (const float*)d_in[10];
  const float* outW    = (const float*)d_in[11];
  const float* out_b   = (const float*)d_in[12];
  float* out = (float*)d_out;

  char* w = (char*)d_ws;
  size_t off = 0;
  auto alloc = [&](size_t bytes) -> char* {
    char* p = w + off;
    off = (off + bytes + 255) & ~(size_t)255;
    return p;
  };

  __bf16* encWihB = (__bf16*)alloc((size_t)GATES4H * EK * 2);
  __bf16* encWhhB = (__bf16*)alloc((size_t)GATES4H * HID * 2);
  __bf16* decWihB = (__bf16*)alloc((size_t)GATES4H * DK * 2);
  __bf16* decWhhB = (__bf16*)alloc((size_t)GATES4H * HID * 2);
  __bf16* outWB   = (__bf16*)alloc((size_t)VOCABP * HID * 2);   // 119.5 MB -> L2-resident
  __bf16* srcEmb  = (__bf16*)alloc((size_t)SRCLEN * BATCH * EK * 2);
  float*  encOut  = (float*) alloc((size_t)BATCH * SRCLEN * HID * 4);
  __bf16* hist    = (__bf16*)alloc((size_t)TSTEPS * BATCH * HID * 2);
  __bf16* xdec    = (__bf16*)alloc((size_t)BATCH * DK * 2);
  // contiguous zero-init region: h ping-pong, hF, cF, ctx, barrier counters
  __bf16* hbf0 = (__bf16*)alloc((size_t)BATCH * HID * 2);
  __bf16* hbf1 = (__bf16*)alloc((size_t)BATCH * HID * 2);
  float*  hF   = (float*) alloc((size_t)BATCH * HID * 4);
  float*  cF   = (float*) alloc((size_t)BATCH * HID * 4);
  float*  ctx  = (float*) alloc((size_t)BATCH * HID * 4);
  unsigned int* barcnt = (unsigned int*)alloc(sizeof(unsigned int) * 128);
  char* zend = w + off;

  {
    int n = (int)((zend - (char*)hbf0) / 4);
    zero_u32_kernel<<<(n + 255) / 256, 256, 0, stream>>>((unsigned int*)hbf0, n);
  }

  auto cvt = [&](const float* s, __bf16* d, int sr, int sc, int dr, int dc) {
    long long n = (long long)dr * dc;
    cvt_pad_kernel<<<(unsigned)((n + 255) / 256), 256, 0, stream>>>(s, d, sr, sc, dr, dc);
  };
  cvt(encWih, encWihB, GATES4H, 250, GATES4H, EK);
  cvt(encWhh, encWhhB, GATES4H, HID, GATES4H, HID);
  cvt(decWih, decWihB, GATES4H, 1274, GATES4H, DK);
  cvt(decWhh, decWhhB, GATES4H, HID, GATES4H, HID);
  cvt(outW,   outWB,   VOCAB,   HID, VOCABP,  HID);

  src_embed_kernel<<<(SRCLEN * BATCH * EK + 255) / 256, 256, 0, stream>>>(src, emb, srcEmb);

  // one persistent kernel for the whole 79-step recurrence (launch-latency bound otherwise)
  recurrence_kernel<<<RBLOCKS, 256, 0, stream>>>(
      srcEmb, encWihB, encWhhB, enc_b, decWihB, decWhhB, dec_b,
      target, emb, hbf0, hbf1, hF, cF, ctx, xdec, hist, encOut, barcnt);

  // one batched projection for all 15 steps (out_W read 3x instead of 15x)
  {
    int waves = (VOCABP / 16) * 3;
    logits_batched_kernel<<<(waves + 7) / 8, 256, 0, stream>>>(hist, outWB, out_b, out);
  }
}
